// DAGLSTM_60361470378541
// MI455X (gfx1250) — matrix operand order repaired and checked by reading, not compiled, not run on hardware
//
#include <hip/hip_runtime.h>

#define DEV __device__ __forceinline__

typedef __attribute__((ext_vector_type(16))) __bf16 v16bf;
typedef __attribute__((ext_vector_type(8)))  float  v8f;

namespace {
constexpr int Bsz  = 16;
constexpr int Nn   = 128;
constexpr int Dd   = 256;
constexpr int Ee   = 1024;
constexpr int Rows = Bsz * Nn;        // 2048
constexpr int INDIM = 3 * Dd + Ee;    // 1792
constexpr float NEGC = 1e30f;
}

// ---------- bf16 helpers ----------

DEV unsigned short bfbits(float f) {           // hardware f32 -> bf16
  union { __bf16 h; unsigned short s; } u;
  u.h = (__bf16)f;
  return u.s;
}
DEV float bff(unsigned short s) {              // bf16 bits -> f32
  union { unsigned u; float f; } x;
  x.u = ((unsigned)s) << 16;
  return x.f;
}

// A-fragment (16xK row-major fp32 source, row stride lda) for 16x16x32 bf16 WMMA.
// lanes 0-15: row=lane,  K = kb*32 + {0..7, 16..23}
// lanes 16-31: row=lane-16, K = kb*32 + {8..15, 24..31}
DEV v16bf afrag_f32(const float* A, int lda, int kb) {
  int lane = threadIdx.x & 31;
  int m   = lane & 15;
  int off = (lane < 16) ? 0 : 8;
  const float* p = A + (size_t)m * lda + kb * 32 + off;
  float4 x0 = *(const float4*)(p);
  float4 x1 = *(const float4*)(p + 4);
  float4 y0 = *(const float4*)(p + 16);
  float4 y1 = *(const float4*)(p + 20);
  union { __bf16 h[16]; v16bf v; } u;
  u.h[0] = (__bf16)x0.x; u.h[1] = (__bf16)x0.y; u.h[2]  = (__bf16)x0.z; u.h[3]  = (__bf16)x0.w;
  u.h[4] = (__bf16)x1.x; u.h[5] = (__bf16)x1.y; u.h[6]  = (__bf16)x1.z; u.h[7]  = (__bf16)x1.w;
  u.h[8] = (__bf16)y0.x; u.h[9] = (__bf16)y0.y; u.h[10] = (__bf16)y0.z; u.h[11] = (__bf16)y0.w;
  u.h[12]= (__bf16)y1.x; u.h[13]= (__bf16)y1.y; u.h[14] = (__bf16)y1.z; u.h[15] = (__bf16)y1.w;
  return u.v;
}

// A-fragment from a pre-built LDS fragment image: two ds_load_b128 per lane.
DEV v16bf afrag_lds(const unsigned short (*F)[32][16], int kb) {
  int lane = threadIdx.x & 31;
  const uint4* p = (const uint4*)&F[kb][lane][0];
  union { uint4 q[2]; v16bf v; } u;
  u.q[0] = p[0]; u.q[1] = p[1];
  return u.v;
}

// Scatter one f32 value into the LDS fragment image at (row m, k = d).
DEV void frag_store(unsigned short (*F)[32][16], int m, int d, float v) {
  int kb = d >> 5, r = d & 31;
  int lane, e;
  if (r < 8)       { lane = m;      e = r;          }
  else if (r < 16) { lane = m + 16; e = r - 8;      }
  else if (r < 24) { lane = m;      e = 8 + r - 16; }
  else             { lane = m + 16; e = 8 + r - 24; }
  F[kb][lane][e] = bfbits(v);
}

// B-fragment from pre-packed weights: 32B contiguous per lane -> two b128 loads.
DEV v16bf bfrag(const unsigned short* P, int tile, int KB, int kb) {
  int lane = threadIdx.x & 31;
  const uint4* p = (const uint4*)(P + ((size_t)(tile * KB + kb) * 32 + lane) * 16);
  union { uint4 q[2]; v16bf v; } u;
  u.q[0] = p[0]; u.q[1] = p[1];
  return u.v;
}

DEV v8f wmma_bf16(v16bf a, v16bf b, v8f c) {
  return __builtin_amdgcn_wmma_f32_16x16x32_bf16(false, a, false, b, (short)0, c,
                                                 false, false);
}

DEV float sigm(float x) { return 1.f / (1.f + __expf(-x)); }

// ---------- weight packing: fp32 (N x K row-major) -> WMMA B-fragments ----------

__global__ void pack_b_kernel(const float* __restrict__ W, unsigned short* __restrict__ out,
                              int Nf, int Kf) {
  int idx = blockIdx.x * blockDim.x + threadIdx.x;
  if (idx >= Nf * Kf) return;
  int e    = idx & 15;
  int lane = (idx >> 4) & 31;
  int blk  = idx >> 9;
  int KB   = Kf >> 5;
  int kb   = blk % KB;
  int tile = blk / KB;
  int n = tile * 16 + (lane & 15);
  int k = kb * 32 + ((lane < 16) ? 0 : 8) + ((e < 8) ? e : e + 8);
  out[idx] = bfbits(W[(size_t)n * Kf + k]);
}

// ---------- generic GEMM: C = act(A @ W^T + bias), one wave per 16x16 tile ----------

__global__ __launch_bounds__(256) void gemm_kernel(const float* __restrict__ A,
                                                   const unsigned short* __restrict__ Bp,
                                                   const float* __restrict__ bias,
                                                   float* __restrict__ C,
                                                   int Mrows, int Ncols, int Kf, int act) {
  int wave = (blockIdx.x * blockDim.x + threadIdx.x) >> 5;
  int ntiles = Ncols >> 4;
  int nwaves = (Mrows >> 4) * ntiles;
  if (wave >= nwaves) return;
  int mt = wave / ntiles, nt = wave % ntiles;
  int KB = Kf >> 5;
  const float* Arow = A + (size_t)mt * 16 * Kf;
  v8f acc = {0.f, 0.f, 0.f, 0.f, 0.f, 0.f, 0.f, 0.f};
  for (int kb = 0; kb < KB; ++kb) {
    v16bf a = afrag_f32(Arow, Kf, kb);
    v16bf b = bfrag(Bp, nt, KB, kb);
    acc = wmma_bf16(a, b, acc);
  }
  int lane = threadIdx.x & 31;
  int col  = nt * 16 + (lane & 15);
  int rb   = mt * 16 + ((lane < 16) ? 0 : 8);
  float bv = bias ? bias[col] : 0.f;
  for (int r = 0; r < 8; ++r) {
    float v = acc[r] + bv;
    if (act) v = fmaxf(v, 0.f);
    C[(size_t)(rb + r) * Ncols + col] = v;
  }
}

// ---------- persistent DAG-LSTM scan: one WGP runs the 128-step recurrence ----------
// Q/M/Ht are staged once per step as bf16 fragment images in LDS (ds_load_b128 in the
// WMMA loops); K2/Vr/H1 state lives in L2-resident scratch. Forget gate is dead.

__global__ __launch_bounds__(1024, 1) void dag_scan_kernel(
    const float* __restrict__ Hl,    // (B*N, D): batch row stride Nn*Dd at fixed step
    const float* __restrict__ qAll,  // (B*N, D) = Hl @ W1^T
    const int*   __restrict__ adj,   // (B, N, N)
    const unsigned short* __restrict__ WcihP, const unsigned short* __restrict__ WchhP,
    const unsigned short* __restrict__ WpihP, const unsigned short* __restrict__ WphhP,
    const unsigned short* __restrict__ W2P,   const unsigned short* __restrict__ WrP,
    const float* __restrict__ bcih, const float* __restrict__ bchh,
    const float* __restrict__ bpih, const float* __restrict__ bphh,
    float* __restrict__ K2, float* __restrict__ Vr, float* __restrict__ H1out) {
  __shared__ float sW[Bsz][Nn];                           // logits -> softmax weights
  __shared__ unsigned short sHc[2][Bsz][Dd];              // per-cell hidden (bf16)
  __shared__ __align__(16) unsigned short sQf[Dd / 32][32][16];   // Q fragment image
  __shared__ __align__(16) unsigned short sMf[Dd / 32][32][16];   // M fragment image
  __shared__ __align__(16) unsigned short sHtf[Dd / 32][32][16];  // Ht fragment image
  const int tid  = threadIdx.x;
  const int lane = tid & 31;
  const int wave = tid >> 5;

  for (int i = 0; i < Nn; ++i) {
    // ---- phase 1: stage Q fragments; attention -> M fragments ----
    for (int t = tid; t < Bsz * Dd; t += 1024) {
      int b = t >> 8, d = t & 255;
      frag_store(sQf, b, d, Hl[((size_t)b * Nn + i) * Dd + d]);
    }
    if (i == 0) {
      for (int t = tid; t < Bsz * Dd; t += 1024) (&sMf[0][0][0])[t] = 0;
    } else {
      for (int t = tid; t < Bsz * Nn; t += 1024) {
        int b = t >> 7, n = t & 127;
        float lg = -NEGC;
        if (n < i) {
          const float* q  = qAll + (size_t)(b * Nn + i) * Dd;
          const float* kv = K2   + (size_t)(b * Nn + n) * Dd;
          float s = 0.f;
          for (int d = 0; d < Dd; d += 4) {
            float4 qa = *(const float4*)(q + d);
            float4 ka = *(const float4*)(kv + d);
            s += qa.x * ka.x + qa.y * ka.y + qa.z * ka.z + qa.w * ka.w;
          }
          float a = (float)adj[((size_t)b * Nn + i) * Nn + n];
          lg = s - (1.f - a) * NEGC;
        }
        sW[b][n] = lg;
      }
      __syncthreads();
      if (wave < Bsz) {  // one wave per batch row: 128-way masked softmax
        int b = wave;
        float v0 = sW[b][lane], v1 = sW[b][lane + 32],
              v2 = sW[b][lane + 64], v3 = sW[b][lane + 96];
        float mx = fmaxf(fmaxf(v0, v1), fmaxf(v2, v3));
        for (int msk = 16; msk >= 1; msk >>= 1) mx = fmaxf(mx, __shfl_xor(mx, msk, 32));
        float e0 = __expf(v0 - mx), e1 = __expf(v1 - mx),
              e2 = __expf(v2 - mx), e3 = __expf(v3 - mx);
        float sm = e0 + e1 + e2 + e3;
        for (int msk = 16; msk >= 1; msk >>= 1) sm += __shfl_xor(sm, msk, 32);
        float inv = 1.f / sm;
        sW[b][lane] = e0 * inv; sW[b][lane + 32] = e1 * inv;
        sW[b][lane + 64] = e2 * inv; sW[b][lane + 96] = e3 * inv;
      }
      __syncthreads();
      for (int t = tid; t < Bsz * Dd; t += 1024) {  // M = w @ Vr -> fragment image
        int b = t >> 8, d = t & 255;
        const float* vb = Vr + (size_t)b * Nn * Dd + d;
        float s = 0.f;
        for (int n = 0; n < i; ++n) {
          if ((n & 7) == 0) __builtin_prefetch(vb + (size_t)(n + 8) * Dd, 0, 1);
          s += sW[b][n] * vb[(size_t)n * Dd];
        }
        frag_store(sMf, b, d, s);
      }
    }
    __syncthreads();

    // ---- phase 2: LSTM gates. wave -> (cell = wave&1, d-tile = wave>>1) ----
    {
      int cell = wave & 1;
      int dt   = wave >> 1;  // 0..15
      const unsigned short* Wih = cell ? WpihP : WcihP;
      const unsigned short* Whh = cell ? WphhP : WchhP;
      v8f aI = {0.f,0.f,0.f,0.f,0.f,0.f,0.f,0.f};
      v8f aG = aI, aO = aI;
#pragma unroll
      for (int kb = 0; kb < Dd / 32; ++kb) {
        v16bf ax = cell ? afrag_lds(sMf, kb) : afrag_lds(sQf, kb);
        v16bf ah = cell ? afrag_lds(sQf, kb) : afrag_lds(sMf, kb);
        v16bf b;
        b = bfrag(Wih, 0 * 16 + dt, 8, kb); aI = wmma_bf16(ax, b, aI);
        b = bfrag(Wih, 2 * 16 + dt, 8, kb); aG = wmma_bf16(ax, b, aG);
        b = bfrag(Wih, 3 * 16 + dt, 8, kb); aO = wmma_bf16(ax, b, aO);
        b = bfrag(Whh, 0 * 16 + dt, 8, kb); aI = wmma_bf16(ah, b, aI);
        b = bfrag(Whh, 2 * 16 + dt, 8, kb); aG = wmma_bf16(ah, b, aG);
        b = bfrag(Whh, 3 * 16 + dt, 8, kb); aO = wmma_bf16(ah, b, aO);
      }
      const float* bi = cell ? bpih : bcih;
      const float* bh = cell ? bphh : bcih ? bchh : bchh;  // (see below, fixed next line)
      bh = cell ? bphh : bchh;
      int c = dt * 16 + (lane & 15);
      float bI = bi[0 * Dd + c] + bh[0 * Dd + c];
      float bG = bi[2 * Dd + c] + bh[2 * Dd + c];
      float bO = bi[3 * Dd + c] + bh[3 * Dd + c];
      int mrow = (lane < 16) ? 0 : 8;
      for (int r = 0; r < 8; ++r) {
        float gi = sigm(aI[r] + bI);
        float gg = tanhf(aG[r] + bG);
        float go = sigm(aO[r] + bO);
        sHc[cell][mrow + r][c] = bfbits(go * tanhf(gi * gg));
      }
    }
    __syncthreads();

    // ---- phase 3: Ht = C + P -> H1 (global f32) and Ht fragment image ----
    for (int t = tid; t < Bsz * Dd; t += 1024) {
      int b = t >> 8, d = t & 255;
      float ht = bff(sHc[0][b][d]) + bff(sHc[1][b][d]);
      H1out[((size_t)b * Nn + i) * Dd + d] = ht;
      frag_store(sHtf, b, d, ht);
    }
    __syncthreads();

    // ---- phase 4: projections K2[:,i] = Ht@W2^T (waves 0-15), Vr[:,i] = Ht@Wr^T ----
    {
      int isv = wave >> 4;
      int dt  = wave & 15;
      const unsigned short* Wp = isv ? WrP : W2P;
      float* Out = isv ? Vr : K2;
      v8f acc = {0.f,0.f,0.f,0.f,0.f,0.f,0.f,0.f};
#pragma unroll
      for (int kb = 0; kb < Dd / 32; ++kb) {
        v16bf a = afrag_lds(sHtf, kb);
        v16bf b = bfrag(Wp, dt, 8, kb);
        acc = wmma_bf16(a, b, acc);
      }
      int col = dt * 16 + (lane & 15);
      int rb  = (lane < 16) ? 0 : 8;
      for (int r = 0; r < 8; ++r)
        Out[((size_t)(rb + r) * Nn + i) * Dd + col] = acc[r];
    }
    __syncthreads();
  }
}

// ---------- small tail kernels ----------

__global__ void concat_kernel(const float* __restrict__ H0, const float* __restrict__ H1,
                              const float* __restrict__ H2, const float* __restrict__ F,
                              float* __restrict__ Hcat) {
  int idx = blockIdx.x * blockDim.x + threadIdx.x;
  if (idx >= Rows * INDIM) return;
  int row = idx / INDIM, c = idx % INDIM;
  float v;
  if (c < Dd)          v = H0[(size_t)row * Dd + c];
  else if (c < 2 * Dd) v = H1[(size_t)row * Dd + c - Dd];
  else if (c < 3 * Dd) v = H2[(size_t)row * Dd + c - 2 * Dd];
  else                 v = F[(size_t)row * Ee + c - 3 * Dd];
  Hcat[idx] = v;
}

__global__ void head_kernel(const float* __restrict__ Hin, const float* __restrict__ W,
                            const float* __restrict__ b, float* __restrict__ out) {
  int idx = blockIdx.x * blockDim.x + threadIdx.x;  // row*7 + cls
  if (idx >= Rows * 7) return;
  int row = idx / 7, c = idx % 7;
  const float* x = Hin + (size_t)row * Dd;
  const float* w = W + (size_t)c * Dd;
  float s = b[c];
  for (int k = 0; k < Dd; ++k) s += x[k] * w[k];
  out[idx] = s;
}

// ---------- host orchestration ----------

extern "C" void kernel_launch(void* const* d_in, const int* in_sizes, int n_in,
                              void* d_out, int out_size, void* d_ws, size_t ws_size,
                              hipStream_t stream) {
  (void)in_sizes; (void)n_in; (void)out_size; (void)ws_size;
  const float* features = (const float*)d_in[0];
  const int*   adj      = (const int*)d_in[1];
  const float* fc1_W = (const float*)d_in[5];
  const float* fc1_b = (const float*)d_in[6];
  const float* W1 = (const float*)d_in[7];
  const float* W2 = (const float*)d_in[8];
  const float* Wr = (const float*)d_in[9];
  const float* Wc_ih = (const float*)d_in[10];
  const float* Wc_hh = (const float*)d_in[11];
  const float* bc_ih = (const float*)d_in[12];
  const float* bc_hh = (const float*)d_in[13];
  const float* Wp_ih = (const float*)d_in[14];
  const float* Wp_hh = (const float*)d_in[15];
  const float* bp_ih = (const float*)d_in[16];
  const float* bp_hh = (const float*)d_in[17];
  const float* m0_W = (const float*)d_in[18];
  const float* m0_b = (const float*)d_in[19];
  const float* m1_W = (const float*)d_in[20];
  const float* m1_b = (const float*)d_in[21];
  const float* m2_W = (const float*)d_in[22];
  const float* m2_b = (const float*)d_in[23];

  char* ws = (char*)d_ws;
  size_t off = 0;
  auto alloc = [&](size_t bytes) -> char* {
    char* p = ws + off;
    off += (bytes + 255) & ~(size_t)255;
    return p;
  };

  unsigned short* fc1P = (unsigned short*)alloc((size_t)Dd * Ee * 2);
  unsigned short *W1P[2], *W2P[2], *WrP[2], *WcihP[2], *WchhP[2], *WpihP[2], *WphhP[2];
  for (int l = 0; l < 2; ++l) {
    W1P[l]   = (unsigned short*)alloc((size_t)Dd * Dd * 2);
    W2P[l]   = (unsigned short*)alloc((size_t)Dd * Dd * 2);
    WrP[l]   = (unsigned short*)alloc((size_t)Dd * Dd * 2);
    WcihP[l] = (unsigned short*)alloc((size_t)4 * Dd * Dd * 2);
    WchhP[l] = (unsigned short*)alloc((size_t)4 * Dd * Dd * 2);
    WpihP[l] = (unsigned short*)alloc((size_t)4 * Dd * Dd * 2);
    WphhP[l] = (unsigned short*)alloc((size_t)4 * Dd * Dd * 2);
  }
  unsigned short* m0P = (unsigned short*)alloc((size_t)Dd * INDIM * 2);
  unsigned short* m1P = (unsigned short*)alloc((size_t)Dd * Dd * 2);

  float* H0   = (float*)alloc((size_t)Rows * Dd * 4);
  float* H1   = (float*)alloc((size_t)Rows * Dd * 4);
  float* H2   = (float*)alloc((size_t)Rows * Dd * 4);
  float* qAll = (float*)alloc((size_t)Rows * Dd * 4);
  float* K2s  = (float*)alloc((size_t)Rows * Dd * 4);
  float* Vrs  = (float*)alloc((size_t)Rows * Dd * 4);
  float* Hcat = (float*)alloc((size_t)Rows * INDIM * 4);
  float* mh0  = (float*)alloc((size_t)Rows * Dd * 4);
  float* mh1  = (float*)alloc((size_t)Rows * Dd * 4);

  auto pack = [&](const float* Wm, unsigned short* out, int N, int K) {
    int total = N * K;
    pack_b_kernel<<<(total + 255) / 256, 256, 0, stream>>>(Wm, out, N, K);
  };
  auto gemm = [&](const float* A, const unsigned short* Bp, const float* bias,
                  float* C, int M, int Nc, int K, int act) {
    int waves = (M / 16) * (Nc / 16);
    gemm_kernel<<<(waves + 7) / 8, 256, 0, stream>>>(A, Bp, bias, C, M, Nc, K, act);
  };

  pack(fc1_W, fc1P, Dd, Ee);
  for (int l = 0; l < 2; ++l) {
    pack(W1 + (size_t)l * Dd * Dd, W1P[l], Dd, Dd);
    pack(W2 + (size_t)l * Dd * Dd, W2P[l], Dd, Dd);
    pack(Wr + (size_t)l * Dd * Dd, WrP[l], Dd, Dd);
    pack(Wc_ih + (size_t)l * 4 * Dd * Dd, WcihP[l], 4 * Dd, Dd);
    pack(Wc_hh + (size_t)l * 4 * Dd * Dd, WchhP[l], 4 * Dd, Dd);
    pack(Wp_ih + (size_t)l * 4 * Dd * Dd, WpihP[l], 4 * Dd, Dd);
    pack(Wp_hh + (size_t)l * 4 * Dd * Dd, WphhP[l], 4 * Dd, Dd);
  }
  pack(m0_W, m0P, Dd, INDIM);
  pack(m1_W, m1P, Dd, Dd);

  gemm(features, fc1P, fc1_b, H0, Rows, Dd, Ee, 1);

  const float* Hin = H0;
  float* Houts[2] = {H1, H2};
  for (int l = 0; l < 2; ++l) {
    gemm(Hin, W1P[l], nullptr, qAll, Rows, Dd, Dd, 0);
    dag_scan_kernel<<<1, 1024, 0, stream>>>(
        Hin, qAll, adj, WcihP[l], WchhP[l], WpihP[l], WphhP[l], W2P[l], WrP[l],
        bc_ih + (size_t)l * 4 * Dd, bc_hh + (size_t)l * 4 * Dd,
        bp_ih + (size_t)l * 4 * Dd, bp_hh + (size_t)l * 4 * Dd,
        K2s, Vrs, Houts[l]);
    Hin = Houts[l];
  }

  concat_kernel<<<(Rows * INDIM + 255) / 256, 256, 0, stream>>>(H0, H1, H2, features, Hcat);
  gemm(Hcat, m0P, m0_b, mh0, Rows, Dd, INDIM, 1);
  gemm(mh0, m1P, m1_b, mh1, Rows, Dd, Dd, 1);
  head_kernel<<<(Rows * 7 + 255) / 256, 256, 0, stream>>>(mh1, m2_W, m2_b, (float*)d_out);
}
